// AxileAttention_73598559585013
// MI455X (gfx1250) — matrix operand order repaired and checked by reading, not compiled, not run on hardware
//
#include <hip/hip_runtime.h>

// ---- CDNA5 WMMA types ----
typedef __attribute__((ext_vector_type(16))) __bf16         v16bf;
typedef __attribute__((ext_vector_type(8)))  float          v8f;
typedef __attribute__((ext_vector_type(8)))  unsigned short u16x8;

struct BfPair { u16x8 lo, hi; };

static __device__ __forceinline__ unsigned short f2bf(float f) {
  union { float f; unsigned u; } v; v.f = f;
  unsigned r = v.u + 0x7FFFu + ((v.u >> 16) & 1u);   // round-to-nearest-even
  return (unsigned short)(r >> 16);
}
static __device__ __forceinline__ unsigned pack2(float a, float b) {
  return (unsigned)f2bf(a) | ((unsigned)f2bf(b) << 16);
}
static __device__ __forceinline__ v16bf mkfrag(u16x8 lo, u16x8 hi) {
  BfPair p{lo, hi};
  return __builtin_bit_cast(v16bf, p);
}
static __device__ __forceinline__ v8f vzero() {
  v8f z = {0.f, 0.f, 0.f, 0.f, 0.f, 0.f, 0.f, 0.f};
  return z;
}

#define CDIM 64
#define HW   256
#define WMAT ((size_t)HW * HW)          // 65536 elems per (s,c) weight matrix
#define SSTR ((size_t)CDIM * WMAT)      // stride between q/k/v weight blocks
#define PANEL 8192                      // 256 n * 32 k elems per K-panel

#define WMMA(acc, A, B) \
  acc = __builtin_amdgcn_wmma_f32_16x16x32_bf16(false, (A), false, (B), (short)0, (acc), false, false)

// B fragment pair (both N-tiles of one wave) = 4 x global_load_b128
struct Bfrag { u16x8 lo0, hi0, lo1, hi1; };
static __device__ __forceinline__ Bfrag loadB(const unsigned short* pb, int bo0, int bo1) {
  Bfrag f;
  f.lo0 = ((const u16x8*)(pb + bo0))[0];
  f.hi0 = ((const u16x8*)(pb + bo0))[1];
  f.lo1 = ((const u16x8*)(pb + bo1))[0];
  f.hi1 = ((const u16x8*)(pb + bo1))[1];
  return f;
}

// =====================================================================
// Prep kernel: fp32 [s][c][k][n] weights -> bf16 panelized Wt[s][c][p][n][k32]
// =====================================================================
__global__ __launch_bounds__(256)
void AxileAttn_prep_weights(const float* __restrict__ qw,
                            const float* __restrict__ kw,
                            const float* __restrict__ vw,
                            unsigned short* __restrict__ wt)
{
  __shared__ unsigned short T[256][40];   // [n][k] panel, 80B row (16B aligned)
  const int tid = threadIdx.x;
  const int p = blockIdx.x;   // K panel 0..7
  const int c = blockIdx.y;   // channel
  const int s = blockIdx.z;   // 0=q 1=k 2=v
  const float* W = (s == 0 ? qw : (s == 1 ? kw : vw)) + (size_t)c * WMAT;

  {
    const int kl = tid >> 3;          // 0..31 local K row
    const int n0 = (tid & 7) << 5;    // col base
    const float4* src = (const float4*)(W + (size_t)((p << 5) + kl) * HW + n0);
#pragma unroll
    for (int i = 0; i < 8; ++i) {
      float4 f = src[i];
      const int n = n0 + 4 * i;
      T[n + 0][kl] = f2bf(f.x);
      T[n + 1][kl] = f2bf(f.y);
      T[n + 2][kl] = f2bf(f.z);
      T[n + 3][kl] = f2bf(f.w);
    }
  }
  __syncthreads();

  unsigned short* dst = wt + ((size_t)(s * CDIM + c) * 8 + p) * PANEL + (size_t)tid * 32;
  const uint4* row = (const uint4*)&T[tid][0];
  uint4* d4 = (uint4*)dst;
  d4[0] = row[0]; d4[1] = row[1]; d4[2] = row[2]; d4[3] = row[3];
}

// =====================================================================
// Main kernel (fast path): 32-row M tile, q/k/v fused in one K loop.
// Per panel/wave: 4 ds_load_b128 (A, shared by all 3 GEMMs),
// 12 global_load_b128 (B), 12 WMMAs; B loads issued one group ahead.
// =====================================================================
__global__ __launch_bounds__(256)
void AxileAttention_73598559585013_kernel(
    const float* __restrict__ x,
    const unsigned short* __restrict__ wt,     // bf16 panelized weights (ws)
    const float* __restrict__ qb, const float* __restrict__ kb, const float* __restrict__ vb,
    float* __restrict__ out)
{
  __shared__ unsigned short Abuf[32][264];   // X tile bf16, [M][K]  (~16.9 KB)
  __shared__ float Sbuf[32][257];            // (~32.9 KB)
  __shared__ float Vbuf[32][257];            // (~32.9 KB)
  __shared__ float Red[32][9];
  __shared__ float RowMax[32];
  __shared__ float RowInvSum[32];

  const int tid = threadIdx.x;
  const int b  = blockIdx.z;
  const int c  = blockIdx.y;
  const int h0 = blockIdx.x << 5;            // 32-row tile

  const size_t bc = (size_t)(b * CDIM + c);
  const float* xp = x   + (bc * HW + h0) * HW;
  float*       op = out + (bc * HW + h0) * HW;

  // ---- Stage X tile (32x256 fp32 -> bf16) into Abuf ----
  {
    const int m  = tid >> 3;
    const int k0 = (tid & 7) << 5;
    const float4* src = (const float4*)(xp + (size_t)m * HW + k0);
    unsigned* dst = (unsigned*)&Abuf[m][k0];
#pragma unroll
    for (int i = 0; i < 8; ++i) {
      float4 f = src[i];
      dst[2 * i]     = pack2(f.x, f.y);
      dst[2 * i + 1] = pack2(f.z, f.w);
    }
  }
  __syncthreads();          // only barrier before the softmax phase

  const int lane  = tid & 31;
  const int wv    = tid >> 5;
  const int half  = lane >> 4;
  const int l16   = lane & 15;
  const int nbase = wv << 5;

  const int col0 = nbase + l16;
  const int col1 = col0 + 16;
  const int koff = half << 4;       // B-layout: lanes 0-15 K=0..15, 16-31 K=16..31
  const int bo0  = col0 * 32 + koff;
  const int bo1  = col1 * 32 + koff;

  const unsigned short* Wq = wt + (size_t)c * WMAT;
  const unsigned short* Wk = Wq + SSTR;
  const unsigned short* Wv = Wk + SSTR;

  v8f Aq[2][2] = {{vzero(), vzero()}, {vzero(), vzero()}};
  v8f Ak[2][2] = {{vzero(), vzero()}, {vzero(), vzero()}};
  v8f Av[2][2] = {{vzero(), vzero()}, {vzero(), vzero()}};

  // pipeline prologue
  Bfrag bq = loadB(Wq, bo0, bo1);

#pragma unroll 2
  for (int p = 0; p < 8; ++p) {
    const int po = p << 13;

    Bfrag bk = loadB(Wk + po, bo0, bo1);     // issued 4 WMMAs ahead of use

    const int ka = (p << 5) + (half << 4);
    const v16bf af0 = mkfrag(*(const u16x8*)&Abuf[l16][ka],
                             *(const u16x8*)&Abuf[l16][ka + 8]);
    const v16bf af1 = mkfrag(*(const u16x8*)&Abuf[16 + l16][ka],
                             *(const u16x8*)&Abuf[16 + l16][ka + 8]);

    {
      const v16bf b0 = mkfrag(bq.lo0, bq.hi0), b1 = mkfrag(bq.lo1, bq.hi1);
      WMMA(Aq[0][0], af0, b0); WMMA(Aq[0][1], af0, b1);
      WMMA(Aq[1][0], af1, b0); WMMA(Aq[1][1], af1, b1);
    }

    Bfrag bv = loadB(Wv + po, bo0, bo1);     // issued 4 WMMAs ahead of use

    {
      const v16bf b0 = mkfrag(bk.lo0, bk.hi0), b1 = mkfrag(bk.lo1, bk.hi1);
      WMMA(Ak[0][0], af0, b0); WMMA(Ak[0][1], af0, b1);
      WMMA(Ak[1][0], af1, b0); WMMA(Ak[1][1], af1, b1);
    }

    if (p < 7) bq = loadB(Wq + po + (1 << 13), bo0, bo1);   // next panel q

    {
      const v16bf b0 = mkfrag(bv.lo0, bv.hi0), b1 = mkfrag(bv.lo1, bv.hi1);
      WMMA(Av[0][0], af0, b0); WMMA(Av[0][1], af0, b1);
      WMMA(Av[1][0], af1, b0); WMMA(Av[1][1], af1, b1);
    }
  }

  // ---- Single epilogue: bias add, s = q*k, stash s and v in LDS ----
  const size_t cb8 = (size_t)c * WMAT + (size_t)h0 * HW;
  const float* qbp = qb + cb8;
  const float* kbp = kb + cb8;
  const float* vbp = vb + cb8;
#pragma unroll
  for (int mt = 0; mt < 2; ++mt) {
#pragma unroll
    for (int r = 0; r < 8; ++r) {
      const int m  = (mt << 4) + r + (half << 3);
      const int o0 = m * HW + col0;
      const int o1 = m * HW + col1;
      const float q0 = Aq[mt][0][r] + qbp[o0], q1 = Aq[mt][1][r] + qbp[o1];
      const float k0 = Ak[mt][0][r] + kbp[o0], k1 = Ak[mt][1][r] + kbp[o1];
      const float v0 = Av[mt][0][r] + vbp[o0], v1 = Av[mt][1][r] + vbp[o1];
      Sbuf[m][col0] = q0 * k0;  Sbuf[m][col1] = q1 * k1;
      Vbuf[m][col0] = v0;       Vbuf[m][col1] = v1;
    }
  }

  __syncthreads();

  // ---- Row softmax over 256 cols (32 rows): two-level LDS reduction ----
  const int rm  = tid >> 3;          // row 0..31
  const int cbx = (tid & 7) << 5;    // 32-elem chunk
  {
    float mx = -3.402823466e38f;
#pragma unroll
    for (int i = 0; i < 32; ++i) mx = fmaxf(mx, Sbuf[rm][cbx + i]);
    Red[rm][tid & 7] = mx;
  }
  __syncthreads();
  if (tid < 32) {
    float m = -3.402823466e38f;
#pragma unroll
    for (int i = 0; i < 8; ++i) m = fmaxf(m, Red[tid][i]);
    RowMax[tid] = m;
  }
  __syncthreads();
  {
    const float rmax = RowMax[rm];
    float sum = 0.f;
#pragma unroll
    for (int i = 0; i < 32; ++i) {
      const float e = __expf(Sbuf[rm][cbx + i] - rmax);
      Sbuf[rm][cbx + i] = e;
      sum += e;
    }
    Red[rm][tid & 7] = sum;
  }
  __syncthreads();
  if (tid < 32) {
    float ssum = 0.f;
#pragma unroll
    for (int i = 0; i < 8; ++i) ssum += Red[tid][i];
    RowInvSum[tid] = 1.0f / ssum;
  }
  __syncthreads();

#pragma unroll 4
  for (int m = 0; m < 32; ++m) {
    op[(size_t)m * HW + tid] = Sbuf[m][tid] * RowInvSum[m] * Vbuf[m][tid];
  }
}

// =====================================================================
// Fallback (self-contained, no workspace): 16-row tile, LDS-staged weights.
// =====================================================================
__global__ __launch_bounds__(256)
void AxileAttention_fallback_kernel(
    const float* __restrict__ x,
    const float* __restrict__ qw, const float* __restrict__ kw, const float* __restrict__ vw,
    const float* __restrict__ qb, const float* __restrict__ kb, const float* __restrict__ vb,
    float* __restrict__ out)
{
  __shared__ unsigned short Abuf[16][264];
  __shared__ unsigned short Bbuf[256][40];
  __shared__ float Sbuf[16][257];
  __shared__ float Vbuf[16][257];
  __shared__ float Red[16][17];
  __shared__ float RowMax[16];
  __shared__ float RowInvSum[16];

  const int tid = threadIdx.x;
  const int b  = blockIdx.z;
  const int c  = blockIdx.y;
  const int h0 = blockIdx.x << 4;

  const size_t bc = (size_t)(b * CDIM + c);
  const float* xp = x   + (bc * HW + h0) * HW;
  float*       op = out + (bc * HW + h0) * HW;

  {
    const int m  = tid >> 4;
    const int k0 = (tid & 15) << 4;
    const float4* src = (const float4*)(xp + (size_t)m * HW + k0);
    unsigned* dst = (unsigned*)&Abuf[m][k0];
#pragma unroll
    for (int i = 0; i < 4; ++i) {
      float4 f = src[i];
      dst[2 * i]     = pack2(f.x, f.y);
      dst[2 * i + 1] = pack2(f.z, f.w);
    }
  }

  const int lane  = tid & 31;
  const int wv    = tid >> 5;
  const int half  = lane >> 4;
  const int l16   = lane & 15;
  const int nbase = wv << 5;

  const float* Ws[3] = {qw, kw, vw};
  const float* Bs[3] = {qb, kb, vb};
  v8f qac0 = vzero(), qac1 = vzero();

  for (int s = 0; s < 3; ++s) {
    const float* Wp = Ws[s] + (size_t)c * WMAT;
    v8f ac0 = vzero(), ac1 = vzero();
    for (int p = 0; p < 8; ++p) {
      __syncthreads();
      {
        const int kl = tid >> 3;
        const int n0 = (tid & 7) << 5;
        const float4* src = (const float4*)(Wp + (size_t)((p << 5) + kl) * HW + n0);
#pragma unroll
        for (int i = 0; i < 8; ++i) {
          float4 f = src[i];
          const int n = n0 + 4 * i;
          Bbuf[n + 0][kl] = f2bf(f.x);
          Bbuf[n + 1][kl] = f2bf(f.y);
          Bbuf[n + 2][kl] = f2bf(f.z);
          Bbuf[n + 3][kl] = f2bf(f.w);
        }
      }
      __syncthreads();
      const u16x8 alo = *(const u16x8*)&Abuf[l16][(p << 5) + (half << 4)];
      const u16x8 ahi = *(const u16x8*)&Abuf[l16][(p << 5) + (half << 4) + 8];
      const v16bf af  = mkfrag(alo, ahi);
      {
        const int col = nbase + l16;
        const u16x8 blo = *(const u16x8*)&Bbuf[col][half << 4];
        const u16x8 bhi = *(const u16x8*)&Bbuf[col][(half << 4) + 8];
        ac0 = __builtin_amdgcn_wmma_f32_16x16x32_bf16(
                  false, af, false, mkfrag(blo, bhi), (short)0, ac0, false, false);
      }
      {
        const int col = nbase + 16 + l16;
        const u16x8 blo = *(const u16x8*)&Bbuf[col][half << 4];
        const u16x8 bhi = *(const u16x8*)&Bbuf[col][(half << 4) + 8];
        ac1 = __builtin_amdgcn_wmma_f32_16x16x32_bf16(
                  false, af, false, mkfrag(blo, bhi), (short)0, ac1, false, false);
      }
    }
    const float* bias = Bs[s] + (size_t)c * WMAT + (size_t)h0 * HW;
#pragma unroll
    for (int r = 0; r < 8; ++r) {
      const int m  = r + (half << 3);
      const int n0 = nbase + l16;
      const int n1 = nbase + 16 + l16;
      const float v0 = ac0[r] + bias[m * HW + n0];
      const float v1 = ac1[r] + bias[m * HW + n1];
      if (s == 0)      { qac0[r] = v0;                 qac1[r] = v1; }
      else if (s == 1) { Sbuf[m][n0] = qac0[r] * v0;   Sbuf[m][n1] = qac1[r] * v1; }
      else             { Vbuf[m][n0] = v0;             Vbuf[m][n1] = v1; }
    }
  }

  __syncthreads();
  const int rm = tid >> 4;
  const int cbx = (tid & 15) << 4;
  {
    float mx = -3.402823466e38f;
#pragma unroll
    for (int i = 0; i < 16; ++i) mx = fmaxf(mx, Sbuf[rm][cbx + i]);
    Red[rm][tid & 15] = mx;
  }
  __syncthreads();
  if (tid < 16) {
    float m = -3.402823466e38f;
#pragma unroll
    for (int i = 0; i < 16; ++i) m = fmaxf(m, Red[tid][i]);
    RowMax[tid] = m;
  }
  __syncthreads();
  {
    const float rmax = RowMax[rm];
    float sum = 0.f;
#pragma unroll
    for (int i = 0; i < 16; ++i) {
      const float e = __expf(Sbuf[rm][cbx + i] - rmax);
      Sbuf[rm][cbx + i] = e;
      sum += e;
    }
    Red[rm][tid & 15] = sum;
  }
  __syncthreads();
  if (tid < 16) {
    float ssum = 0.f;
#pragma unroll
    for (int i = 0; i < 16; ++i) ssum += Red[tid][i];
    RowInvSum[tid] = 1.0f / ssum;
  }
  __syncthreads();
#pragma unroll 4
  for (int m = 0; m < 16; ++m) {
    op[(size_t)m * HW + tid] = Sbuf[m][tid] * RowInvSum[m] * Vbuf[m][tid];
  }
}

extern "C" void kernel_launch(void* const* d_in, const int* in_sizes, int n_in,
                              void* d_out, int out_size, void* d_ws, size_t ws_size,
                              hipStream_t stream) {
  (void)in_sizes; (void)n_in; (void)out_size;
  const float* x  = (const float*)d_in[0];
  const float* qw = (const float*)d_in[1];
  const float* kw = (const float*)d_in[2];
  const float* vw = (const float*)d_in[3];
  const float* qb = (const float*)d_in[4];
  const float* kb = (const float*)d_in[5];
  const float* vb = (const float*)d_in[6];
  float* out = (float*)d_out;

  const size_t ws_need = (size_t)3 * CDIM * WMAT * sizeof(unsigned short); // 24 MB
  if (d_ws != nullptr && ws_size >= ws_need) {
    unsigned short* wt = (unsigned short*)d_ws;
    hipLaunchKernelGGL(AxileAttn_prep_weights, dim3(8, CDIM, 3), dim3(256), 0, stream,
                       qw, kw, vw, wt);
    hipLaunchKernelGGL(AxileAttention_73598559585013_kernel, dim3(8, CDIM, 8), dim3(256), 0, stream,
                       x, wt, qb, kb, vb, out);
  } else {
    hipLaunchKernelGGL(AxileAttention_fallback_kernel, dim3(16, CDIM, 8), dim3(256), 0, stream,
                       x, qw, kw, vw, qb, kb, vb, out);
  }
}
